// QNNAlphaDetailed_64622077935673
// MI455X (gfx1250) — compile-verified, matrix operand order checked by reading, not run on hardware
//
#include <hip/hip_runtime.h>
#include <stdint.h>

typedef __attribute__((ext_vector_type(16))) __bf16 v16bf;
typedef __attribute__((ext_vector_type(8)))  float  v8f;
typedef __attribute__((ext_vector_type(4)))  int    v4i;

union Frag { v16bf v; uint4 u[2]; unsigned short s[16]; };

__device__ __forceinline__ unsigned short f2bf(float f) {
  unsigned int u = __float_as_uint(f);
  unsigned int r = (u + 0x7FFFu + ((u >> 16) & 1u)) >> 16;
  return (unsigned short)r;
}
__device__ __forceinline__ float bf2f(unsigned short h) {
  return __uint_as_float(((unsigned int)h) << 16);
}

// ---- CDNA5 async global->LDS staging (ASYNCcnt path), with compile-safe fallback ----
#if __has_builtin(__builtin_amdgcn_global_load_async_to_lds_b128)
#define HAS_ASYNC_LDS 1
#else
#define HAS_ASYNC_LDS 0
#endif

#if HAS_ASYNC_LDS
typedef __attribute__((address_space(1))) void as1_void;
typedef __attribute__((address_space(3))) void as3_void;
typedef __attribute__((address_space(1))) v4i* as1_v4i_ptr;
typedef __attribute__((address_space(3))) v4i* as3_v4i_ptr;
__device__ __forceinline__ void async_copy16(const unsigned short* g, unsigned short* l) {
  __builtin_amdgcn_global_load_async_to_lds_b128((as1_v4i_ptr)(as1_void*)g,
                                                 (as3_v4i_ptr)(as3_void*)l, 0, 0);
}
template <int N> __device__ __forceinline__ void wait_async() {
  if constexpr (N == 0)      asm volatile("s_wait_asynccnt 0x0" ::: "memory");
  else if constexpr (N == 1) asm volatile("s_wait_asynccnt 0x1" ::: "memory");
  else                       asm volatile("s_wait_asynccnt 0x2" ::: "memory");
}
#endif

// ---------------- prep: fp32 [batch][rows][cols] -> bf16 [batch][cols][rows] ----------------
__global__ void k_transpose_bf16(const float* __restrict__ src, unsigned short* __restrict__ dst,
                                 int batch, int rows, int cols) {
  int total = batch * rows * cols;
  int idx = blockIdx.x * blockDim.x + threadIdx.x;
  if (idx >= total) return;
  int bi = idx / (rows * cols);
  int rem = idx % (rows * cols);
  int r = rem / cols, c = rem % cols;
  dst[(size_t)bi * rows * cols + (size_t)c * rows + r] = f2bf(src[idx]);
}

// ---------------- RMSNorm: z -> bf16 into X[:, 0:2048] ----------------
__global__ void __launch_bounds__(256)
k_rmsnorm(const float* __restrict__ feats, const float* __restrict__ wn,
          unsigned short* __restrict__ X) {
  __shared__ float red[256];
  int row = blockIdx.x;
  int tid = threadIdx.x;
  const float* f = feats + (size_t)row * 2048;
  float vals[8]; float ss = 0.f;
  #pragma unroll
  for (int k = 0; k < 8; ++k) { float v = f[tid + k * 256]; vals[k] = v; ss += v * v; }
  red[tid] = ss; __syncthreads();
  for (int s = 128; s > 0; s >>= 1) { if (tid < s) red[tid] += red[tid + s]; __syncthreads(); }
  float scale = rsqrtf(red[0] * (1.0f / 2048.0f) + 1e-6f);
  unsigned short* xr = X + (size_t)row * 4096;
  #pragma unroll
  for (int k = 0; k < 8; ++k) { int i = tid + k * 256; xr[i] = f2bf(vals[k] * wn[i] * scale); }
}

// ---------------- fused FM interaction (WMMA bf16) ----------------
// block = 256 threads (8 waves); wave w = head w; 16 batch rows per block.
__global__ void __launch_bounds__(256)
k_interaction(const unsigned short* __restrict__ X, const unsigned short* __restrict__ Ut,
              const unsigned short* __restrict__ Vt, float* __restrict__ inter) {
  __shared__ float qs[8 * 16 * 32];     // per-wave qsum [16 b][32 r]
  const int tid  = threadIdx.x;
  const int lane = tid & 31;
  const int w    = tid >> 5;            // head index
  const int bBase = blockIdx.x * 16;
  const int ln = lane & 15;
  const int hi = lane >> 4;
  const int c  = hi * 8;

  // Preload U (B operand) tiles: Ut[h][r=32][d=32] bf16, K = d
  Frag bu[2];
  #pragma unroll
  for (int t = 0; t < 2; ++t) {
    const unsigned short* p = Ut + ((size_t)(w * 32 + t * 16 + ln)) * 32;
    bu[t].u[0] = *(const uint4*)(p + c);
    bu[t].u[1] = *(const uint4*)(p + 16 + c);
  }

  for (int b = 0; b < 16; ++b) {
    const unsigned short* zr = X + (size_t)(bBase + b) * 4096;   // z half, [f=64][d=32]
    #pragma unroll
    for (int t = 0; t < 2; ++t) {       // r tile (0..15 / 16..31)
      float S[3] = {0.f, 0.f, 0.f}, SS[3] = {0.f, 0.f, 0.f};
      #pragma unroll
      for (int i = 0; i < 4; ++i) {     // f tiles; blocks = {0},{1,2},{3}
        Frag a;
        const unsigned short* zrow = zr + (size_t)(i * 16 + ln) * 32;
        a.u[0] = *(const uint4*)(zrow + c);
        a.u[1] = *(const uint4*)(zrow + 16 + c);
        v8f acc = {};
        acc = __builtin_amdgcn_wmma_f32_16x16x32_bf16(false, a.v, false, bu[t].v,
                                                      (short)0, acc, false, false);
        float ps = 0.f, pss = 0.f;
        #pragma unroll
        for (int j = 0; j < 8; ++j) { float v = acc[j]; ps += v; pss += v * v; }
        ps  += __shfl_xor(ps,  16, 32);   // combine M-halves (lane L and L+16)
        pss += __shfl_xor(pss, 16, 32);
        int blk = (i == 0) ? 0 : ((i == 3) ? 2 : 1);
        S[blk] += ps; SS[blk] += pss;
      }
      float q = 0.f;
      #pragma unroll
      for (int k = 0; k < 3; ++k) q += S[k] * S[k] - SS[k];
      if (hi == 0) qs[w * 512 + b * 32 + t * 16 + ln] = q;
    }
  }
  __syncthreads();

  // Projection: qsum (16 b x 32 r) @ Vt[h] (256 p x 32 r, K-major) -> inter
  Frag aq;
  {
    const float* qrow = qs + w * 512 + ln * 32;   // row m = b = ln
    #pragma unroll
    for (int e = 0; e < 8; ++e) aq.s[e]     = f2bf(qrow[c + e]);
    #pragma unroll
    for (int e = 0; e < 8; ++e) aq.s[8 + e] = f2bf(qrow[16 + c + e]);
  }
  #pragma unroll 4
  for (int nt = 0; nt < 16; ++nt) {
    Frag bv;
    const unsigned short* vp = Vt + ((size_t)(w * 256 + nt * 16 + ln)) * 32;
    bv.u[0] = *(const uint4*)(vp + c);
    bv.u[1] = *(const uint4*)(vp + 16 + c);
    v8f acc = {};
    acc = __builtin_amdgcn_wmma_f32_16x16x32_bf16(false, aq.v, false, bv.v,
                                                  (short)0, acc, false, false);
    #pragma unroll
    for (int j = 0; j < 8; ++j) {
      int m = j + hi * 8;               // batch row within tile
      inter[(size_t)(bBase + m) * 2048 + w * 256 + nt * 16 + ln] = acc[j];
    }
  }
}

// ---------------- SE: column mean over batch ----------------
__global__ void k_colmean(const float* __restrict__ inter, float* __restrict__ mbuf) {
  int c = blockIdx.x * blockDim.x + threadIdx.x;   // 0..2047
  float s = 0.f;
  for (int b = 0; b < 8192; ++b) s += inter[(size_t)b * 2048 + c];
  mbuf[c] = s * (1.0f / 8192.0f);
}

// ---------------- SE: tiny MLP -> sigmoid scale vector ----------------
__global__ void k_se_mlp(const float* __restrict__ mbuf,
                         const float* __restrict__ w1, const float* __restrict__ b1,
                         const float* __restrict__ w2, const float* __restrict__ b2,
                         float* __restrict__ wbuf) {
  __shared__ float hid[256];
  int t = threadIdx.x;
  float h = b1[t];
  for (int i = 0; i < 2048; ++i) h += mbuf[i] * w1[i * 256 + t];
  hid[t] = fmaxf(h, 0.f);
  __syncthreads();
  for (int cidx = t; cidx < 2048; cidx += 256) {
    float a = b2[cidx];
    for (int j = 0; j < 256; ++j) a += hid[j] * w2[j * 2048 + cidx];
    wbuf[cidx] = 1.f / (1.f + expf(-a));
  }
}

// ---------------- scale inter by SE weights -> bf16 into X[:, 2048:4096] ----------------
__global__ void k_apply_scale(const float* __restrict__ inter, const float* __restrict__ wbuf,
                              unsigned short* __restrict__ X) {
  int idx = blockIdx.x * blockDim.x + threadIdx.x;   // < 8192*2048
  int b = idx >> 11, cidx = idx & 2047;
  X[(size_t)b * 4096 + 2048 + cidx] = f2bf(inter[idx] * wbuf[cidx]);
}

// ---------------- WMMA bf16 GEMM with LDS-staged (async) A panel ----------------
// out = relu(A[M,K] @ Bt[N,K]^T + bias), bf16 out.
// Block = 8 waves. Wave -> (mSub, nGroup); each wave computes 32x64.
// A panel (ROWS x 64 per chunk) staged once per block via GLOBAL_LOAD_ASYNC_TO_LDS_B128,
// double-buffered with s_wait_asynccnt + barrier.
template <int MSUBS>
__global__ void __launch_bounds__(256)
k_gemm_tile(const unsigned short* __restrict__ A, const unsigned short* __restrict__ Bt,
            const float* __restrict__ bias, unsigned short* __restrict__ out,
            int M, int N, int K) {
  constexpr int ROWS = MSUBS * 32;
  __shared__ unsigned short abuf[2][ROWS * 64];

  const int tid  = threadIdx.x;
  const int lane = tid & 31;
  const int w    = tid >> 5;
  const int nGroups = 8 / MSUBS;
  const int mSub   = w / nGroups;
  const int nGroup = w % nGroups;
  const int ln = lane & 15;
  const int hi = lane >> 4;
  const int c  = hi * 8;
  const int mBlockBase = blockIdx.x * ROWS;
  const int nBase = nGroup * 64;

  v8f acc[2][4];
  #pragma unroll
  for (int mt = 0; mt < 2; ++mt)
    #pragma unroll
    for (int nb = 0; nb < 4; ++nb) acc[mt][nb] = v8f{};

  const unsigned short* brow[4];
  #pragma unroll
  for (int nb = 0; nb < 4; ++nb) brow[nb] = Bt + (size_t)(nBase + nb * 16 + ln) * K;

  auto stage = [&](int kk, int buf) {
    #pragma unroll
    for (int t = 0; t < MSUBS; ++t) {          // ROWS*64 bf16 = ROWS*8 x 16B; 256 threads
      int idx = tid + t * 256;
      int row = idx >> 3;
      int off = (idx & 7) * 8;
      const unsigned short* g = A + (size_t)(mBlockBase + row) * K + kk + off;
      unsigned short* l = &abuf[buf][row * 64 + off];
#if HAS_ASYNC_LDS
      async_copy16(g, l);
#else
      *(uint4*)l = *(const uint4*)g;
#endif
    }
  };

  auto compute = [&](int kk, int buf) {
    #pragma unroll
    for (int ks = 0; ks < 2; ++ks) {
      const int kl = ks * 32;
      Frag af[2];
      #pragma unroll
      for (int mt = 0; mt < 2; ++mt) {
        const unsigned short* p = &abuf[buf][(mSub * 32 + mt * 16 + ln) * 64 + kl];
        af[mt].u[0] = *(const uint4*)(p + c);
        af[mt].u[1] = *(const uint4*)(p + 16 + c);
      }
      #pragma unroll
      for (int nb = 0; nb < 4; ++nb) {
        Frag bfr;
        const unsigned short* bp = brow[nb] + kk + kl;
        bfr.u[0] = *(const uint4*)(bp + c);
        bfr.u[1] = *(const uint4*)(bp + 16 + c);
        #pragma unroll
        for (int mt = 0; mt < 2; ++mt)
          acc[mt][nb] = __builtin_amdgcn_wmma_f32_16x16x32_bf16(false, af[mt].v, false, bfr.v,
                                                                (short)0, acc[mt][nb], false, false);
      }
    }
  };

  const int nChunks = K >> 6;
#if HAS_ASYNC_LDS
  stage(0, 0);
  for (int ch = 0; ch < nChunks; ++ch) {
    if (ch + 1 < nChunks) {
      stage((ch + 1) << 6, (ch + 1) & 1);   // buffer of chunk ch-1: compute done (barrier)
      wait_async<MSUBS>();                  // chunk ch complete (per-wave), next in flight
    } else {
      wait_async<0>();
    }
    __syncthreads();                        // panel visible to all waves
    compute(ch << 6, ch & 1);
    __syncthreads();                        // compute done before buffer reuse
  }
#else
  for (int ch = 0; ch < nChunks; ++ch) {
    __syncthreads();
    stage(ch << 6, 0);
    __syncthreads();
    compute(ch << 6, 0);
  }
#endif

  #pragma unroll
  for (int mt = 0; mt < 2; ++mt)
    #pragma unroll
    for (int nb = 0; nb < 4; ++nb) {
      int n = nBase + nb * 16 + ln;
      float bv = bias[n];
      #pragma unroll
      for (int j = 0; j < 8; ++j) {
        int m = mBlockBase + mSub * 32 + mt * 16 + j + hi * 8;
        out[(size_t)m * N + n] = f2bf(fmaxf(acc[mt][nb][j] + bv, 0.f));
      }
    }
  (void)M;
}

// ---------------- final 256-dot per row ----------------
__global__ void k_final(const unsigned short* __restrict__ H2, const float* __restrict__ W3,
                        const float* __restrict__ b3, float* __restrict__ out) {
  int tid = threadIdx.x;
  int lane = tid & 31;
  int row = blockIdx.x * 8 + (tid >> 5);
  uint4 u = *(const uint4*)(H2 + (size_t)row * 256 + lane * 8);
  const unsigned short* us = (const unsigned short*)&u;
  float s = 0.f;
  #pragma unroll
  for (int k = 0; k < 8; ++k) s += bf2f(us[k]) * W3[lane * 8 + k];
  #pragma unroll
  for (int off = 16; off > 0; off >>= 1) s += __shfl_xor(s, off, 32);
  if (lane == 0) out[row] = s + b3[0];
}

extern "C" void kernel_launch(void* const* d_in, const int* in_sizes, int n_in,
                              void* d_out, int out_size, void* d_ws, size_t ws_size,
                              hipStream_t stream) {
  const float* feats = (const float*)d_in[0];
  const float* wnorm = (const float*)d_in[1];
  const float* U     = (const float*)d_in[2];
  const float* V     = (const float*)d_in[3];
  const float* sew1  = (const float*)d_in[4];
  const float* seb1  = (const float*)d_in[5];
  const float* sew2  = (const float*)d_in[6];
  const float* seb2  = (const float*)d_in[7];
  const float* W1    = (const float*)d_in[8];
  const float* b1    = (const float*)d_in[9];
  const float* W2    = (const float*)d_in[10];
  const float* b2    = (const float*)d_in[11];
  const float* W3    = (const float*)d_in[12];
  const float* b3    = (const float*)d_in[13];
  float* out = (float*)d_out;

  char* ws = (char*)d_ws;
  size_t off = 0;
  auto alloc = [&](size_t bytes) -> void* {
    void* p = ws + off; off = (off + bytes + 255) & ~(size_t)255; return p;
  };
  unsigned short* X     = (unsigned short*)alloc((size_t)8192 * 4096 * 2); // [z | inter*w] bf16
  float*          inter = (float*)         alloc((size_t)8192 * 2048 * 4);
  unsigned short* W1t   = (unsigned short*)alloc((size_t)4096 * 512 * 2);
  unsigned short* W2t   = (unsigned short*)alloc((size_t)512 * 256 * 2);
  unsigned short* Utb   = (unsigned short*)alloc((size_t)8 * 32 * 32 * 2);
  unsigned short* Vtb   = (unsigned short*)alloc((size_t)8 * 32 * 256 * 2);
  unsigned short* H1    = (unsigned short*)alloc((size_t)8192 * 512 * 2);
  unsigned short* H2    = (unsigned short*)alloc((size_t)8192 * 256 * 2);
  float* mbuf = (float*)alloc(2048 * 4);
  float* wbuf = (float*)alloc(2048 * 4);

  // weight prep (bf16, K-major for WMMA B operands)
  k_transpose_bf16<<<(8 * 32 * 32 + 255) / 256, 256, 0, stream>>>(U, Utb, 8, 32, 32);
  k_transpose_bf16<<<(8 * 32 * 256 + 255) / 256, 256, 0, stream>>>(V, Vtb, 8, 32, 256);
  k_transpose_bf16<<<(4096 * 512 + 255) / 256, 256, 0, stream>>>(W1, W1t, 1, 4096, 512);
  k_transpose_bf16<<<(512 * 256 + 255) / 256, 256, 0, stream>>>(W2, W2t, 1, 512, 256);

  k_rmsnorm<<<8192, 256, 0, stream>>>(feats, wnorm, X);
  k_interaction<<<512, 256, 0, stream>>>(X, Utb, Vtb, inter);
  k_colmean<<<8, 256, 0, stream>>>(inter, mbuf);
  k_se_mlp<<<1, 256, 0, stream>>>(mbuf, sew1, seb1, sew2, seb2, wbuf);
  k_apply_scale<<<65536, 256, 0, stream>>>(inter, wbuf, X);
  k_gemm_tile<1><<<8192 / 32, 256, 0, stream>>>(X, W1t, b1, H1, 8192, 512, 4096);
  k_gemm_tile<2><<<8192 / 64, 256, 0, stream>>>(H1, W2t, b2, H2, 8192, 256, 512);
  k_final<<<1024, 256, 0, stream>>>(H2, W3, b3, out);

  (void)in_sizes; (void)n_in; (void)out_size; (void)ws_size;
}
